// Attention_66632122630620
// MI455X (gfx1250) — matrix operand order repaired and checked by reading, not compiled, not run on hardware
//
#include <hip/hip_runtime.h>
#include <hip/hip_bf16.h>

// ---- CDNA5 (gfx1250) bf16 WMMA pipeline for fused per-token attention ----
// Stages (all on `stream`, sequential):
//   0) cast x -> bf16 ; transpose+cast W_qkv, W_out -> bf16 (N-major for B frags)
//   1) qkv  = x_bf16 @ Wqkv^T   (WMMA bf16, f32 accum, bf16 out)
//   2) per-token head-mix attention (WMMA bf16 + lane-shuffle softmax)
//   3) out  = attn_out @ Wout^T + b   (WMMA bf16, f32 out)
// GEMM wave tile is 64x64 (4x4 WMMA frags): 16 WMMAs per 8 fragment loads,
// doubling compute-per-cache-byte vs a 32x64 tile (workload is compute-bound).

typedef __attribute__((ext_vector_type(16))) __bf16 v16bf;
typedef __attribute__((ext_vector_type(8)))  __bf16 v8bf;
typedef __attribute__((ext_vector_type(8)))  float  v8f;

#define WMMA_BF16(a, b, c) \
  __builtin_amdgcn_wmma_f32_16x16x32_bf16(false, (a), false, (b), (short)0, (c), false, false)

// A fragment, 16x32 bf16, row-major source (ld in elements).
// lane&15 = row M; lanes 0-15 hold K in {0..7, 16..23}, lanes 16-31 {8..15, 24..31}.
__device__ __forceinline__ v16bf load_fragA(const __bf16* __restrict__ base, int ld) {
  const int lane = threadIdx.x & 31;
  const int r  = lane & 15;
  const int kb = (lane >> 4) << 3;               // 0 or 8
  const __bf16* p = base + (size_t)r * ld + kb;
  v8bf lo = *(const v8bf*)(p);                   // K = kb .. kb+7
  v8bf hi = *(const v8bf*)(p + 16);              // K = 16+kb .. 16+kb+7
  v16bf f;
#pragma unroll
  for (int i = 0; i < 8; ++i) { f[i] = lo[i]; f[8 + i] = hi[i]; }
  return f;
}

// B fragment, 32x16 bf16, from N-major storage Bt[n][k] (ld = K stride).
// lane&15 = column N; lanes 0-15 hold K in {0..7, 16..23}, lanes 16-31 {8..15, 24..31}.
__device__ __forceinline__ v16bf load_fragB(const __bf16* __restrict__ base, int ld) {
  const int lane = threadIdx.x & 31;
  const int n  = lane & 15;
  const int ks = (lane >> 4) << 3;               // 0 or 8
  const __bf16* p = base + (size_t)n * ld + ks;
  v8bf lo = *(const v8bf*)(p);
  v8bf hi = *(const v8bf*)(p + 16);
  v16bf f;
#pragma unroll
  for (int i = 0; i < 8; ++i) { f[i] = lo[i]; f[8 + i] = hi[i]; }
  return f;
}

// ---------------- stage 0: conversions ----------------
__global__ void k_cast_bf16(const float* __restrict__ in, __bf16* __restrict__ out, int n) {
  int i = blockIdx.x * blockDim.x + threadIdx.x;
  int stride = gridDim.x * blockDim.x;
  for (; i < n; i += stride) out[i] = (__bf16)in[i];
}

// W is [K x N] row-major (row stride N); Wt is [N x K] bf16.
__global__ void k_transpose_bf16(const float* __restrict__ W, __bf16* __restrict__ Wt,
                                 int N, int K) {
  int i = blockIdx.x * blockDim.x + threadIdx.x;
  int total = N * K;
  int stride = gridDim.x * blockDim.x;
  for (; i < total; i += stride) {
    int n = i / K;
    int k = i - n * K;
    Wt[i] = (__bf16)W[(size_t)k * N + n];
  }
}

// ---------------- GEMM: C[bf16] = A @ Bt^T ----------------
// Block = 256M x 128N, 8 waves (4x2), wave tile 64M x 64N (4x4 WMMA frags).
__global__ void __launch_bounds__(256)
k_gemm_bf16out(const __bf16* __restrict__ A, const __bf16* __restrict__ Bt,
               __bf16* __restrict__ C, int K, int ldc) {
  const int wave = threadIdx.x >> 5;
  const int lane = threadIdx.x & 31;
  const int m0 = blockIdx.y * 256 + (wave >> 1) * 64;
  const int n0 = blockIdx.x * 128 + (wave & 1) * 64;

  v8f acc[4][4];
#pragma unroll
  for (int i = 0; i < 4; ++i)
#pragma unroll
    for (int j = 0; j < 4; ++j) acc[i][j] = (v8f){};

#pragma unroll 2
  for (int k = 0; k < K; k += 32) {
    v16bf a[4], b[4];
#pragma unroll
    for (int i = 0; i < 4; ++i)
      a[i] = load_fragA(A + (size_t)(m0 + 16 * i) * K + k, K);
#pragma unroll
    for (int j = 0; j < 4; ++j)
      b[j] = load_fragB(Bt + (size_t)(n0 + 16 * j) * K + k, K);
#pragma unroll
    for (int i = 0; i < 4; ++i)
#pragma unroll
      for (int j = 0; j < 4; ++j)
        acc[i][j] = WMMA_BF16(a[i], b[j], acc[i][j]);
  }

  const int hi = lane >> 4, nl = lane & 15;
#pragma unroll
  for (int i = 0; i < 4; ++i)
#pragma unroll
    for (int j = 0; j < 4; ++j)
#pragma unroll
      for (int r = 0; r < 8; ++r)
        C[(size_t)(m0 + 16 * i + r + 8 * hi) * ldc + (n0 + 16 * j + nl)] =
            (__bf16)acc[i][j][r];
}

// ---------------- GEMM: C[f32] = A @ Bt^T + bias ----------------
__global__ void __launch_bounds__(256)
k_gemm_f32out_bias(const __bf16* __restrict__ A, const __bf16* __restrict__ Bt,
                   const float* __restrict__ bias, float* __restrict__ C,
                   int K, int ldc) {
  const int wave = threadIdx.x >> 5;
  const int lane = threadIdx.x & 31;
  const int m0 = blockIdx.y * 256 + (wave >> 1) * 64;
  const int n0 = blockIdx.x * 128 + (wave & 1) * 64;

  v8f acc[4][4];
#pragma unroll
  for (int i = 0; i < 4; ++i)
#pragma unroll
    for (int j = 0; j < 4; ++j) acc[i][j] = (v8f){};

#pragma unroll 2
  for (int k = 0; k < K; k += 32) {
    v16bf a[4], b[4];
#pragma unroll
    for (int i = 0; i < 4; ++i)
      a[i] = load_fragA(A + (size_t)(m0 + 16 * i) * K + k, K);
#pragma unroll
    for (int j = 0; j < 4; ++j)
      b[j] = load_fragB(Bt + (size_t)(n0 + 16 * j) * K + k, K);
#pragma unroll
    for (int i = 0; i < 4; ++i)
#pragma unroll
      for (int j = 0; j < 4; ++j)
        acc[i][j] = WMMA_BF16(a[i], b[j], acc[i][j]);
  }

  const int hi = lane >> 4, nl = lane & 15;
#pragma unroll
  for (int j = 0; j < 4; ++j) {
    const float bv = bias[n0 + 16 * j + nl];
#pragma unroll
    for (int i = 0; i < 4; ++i)
#pragma unroll
      for (int r = 0; r < 8; ++r)
        C[(size_t)(m0 + 16 * i + r + 8 * hi) * ldc + (n0 + 16 * j + nl)] =
            acc[i][j][r] + bv;
  }
}

// ---------------- per-token head-mix attention ----------------
// One wave32 per token. q,k,v are [16 heads x 64 dims] bf16 views of a qkv row.
__global__ void __launch_bounds__(256)
k_attention(const __bf16* __restrict__ qkv, __bf16* __restrict__ aout) {
  __shared__ __bf16 lds_attn[8][16][16];   // per-wave attn staging (bf16)
  const int wave = threadIdx.x >> 5;
  const int lane = threadIdx.x & 31;
  const int t = blockIdx.x * 8 + wave;

  const __bf16* q = qkv + (size_t)t * 3072;
  const __bf16* kmat = q + 1024;
  const __bf16* vmat = q + 2048;

  // scores[h][g] = sum_d q[h][d] * k[g][d]  (K = 64 -> two K=32 WMMAs)
  v8f s = (v8f){};
#pragma unroll
  for (int c = 0; c < 64; c += 32) {
    v16bf aq = load_fragA(q + c, 64);
    v16bf bk = load_fragB(kmat + c, 64);   // B = k^T: N-major storage == k rows
    s = WMMA_BF16(aq, bk, s);
  }

  // softmax over g (the N axis => across each 16-lane half), per C component.
  const int hi = lane >> 4, nl = lane & 15;
#pragma unroll
  for (int r = 0; r < 8; ++r) {
    float x = s[r] * 0.125f;               // 1/sqrt(64)
    float m = x;
    m = fmaxf(m, __shfl_xor(m, 1, 16));
    m = fmaxf(m, __shfl_xor(m, 2, 16));
    m = fmaxf(m, __shfl_xor(m, 4, 16));
    m = fmaxf(m, __shfl_xor(m, 8, 16));
    float e = __expf(x - m);
    float sum = e;
    sum += __shfl_xor(sum, 1, 16);
    sum += __shfl_xor(sum, 2, 16);
    sum += __shfl_xor(sum, 4, 16);
    sum += __shfl_xor(sum, 8, 16);
    s[r] = e / sum;
  }

  // C layout (lane=N, vgpr=M) -> row-major LDS so we can regather as A frag.
#pragma unroll
  for (int r = 0; r < 8; ++r)
    lds_attn[wave][r + 8 * hi][nl] = (__bf16)s[r];
  __syncthreads();

  // out[h][d] = sum_g attn[h][g] * v[g][d];  K=16 zero-padded to 32.
  const int kb = hi << 3;
  const bool hiHalf = hi != 0;
#pragma unroll
  for (int nt = 0; nt < 4; ++nt) {
    v16bf af, bv;
    const int col = nt * 16 + nl;
#pragma unroll
    for (int e = 0; e < 8; ++e) {
      af[e]     = lds_attn[wave][nl][kb + e];            // K = kb+e (< 16)
      af[8 + e] = (__bf16)0.0f;                          // K >= 16 -> 0
    }
#pragma unroll
    for (int e = 0; e < 16; ++e) {
      // element e<8 holds K=kb+e (valid); e>=8 holds K>=16 -> 0
      __bf16 val = vmat[(size_t)(kb + (e & 7)) * 64 + col];
      bv[e] = (e >= 8) ? (__bf16)0.0f : val;
    }
    v8f z = (v8f){};
    v8f o = WMMA_BF16(af, bv, z);
#pragma unroll
    for (int r = 0; r < 8; ++r)
      aout[(size_t)t * 1024 + (size_t)(r + 8 * hi) * 64 + col] = (__bf16)o[r];
  }
}

// ---------------- launch ----------------
extern "C" void kernel_launch(void* const* d_in, const int* in_sizes, int n_in,
                              void* d_out, int out_size, void* d_ws, size_t ws_size,
                              hipStream_t stream) {
  const float* x     = (const float*)d_in[0];   // [4,4096,1024]
  const float* Wqkv  = (const float*)d_in[1];   // [1024,3072]
  const float* Wout  = (const float*)d_in[2];   // [1024,1024]
  const float* bout  = (const float*)d_in[3];   // [1024]
  float* out = (float*)d_out;                   // [4,4096,1024]

  const int T = 16384;           // 4*4096 tokens
  char* ws = (char*)d_ws;
  // workspace layout (x_bf region reused for attn_out after GEMM1 consumes it)
  __bf16* x_bf   = (__bf16*)(ws);                                   // 32 MB
  __bf16* wqkv_t = (__bf16*)(ws + (size_t)33554432);                //  6 MB
  __bf16* wout_t = (__bf16*)(ws + (size_t)33554432 + 6291456);      //  2 MB
  __bf16* qkv    = (__bf16*)(ws + (size_t)33554432 + 6291456 + 2097152); // 100 MB
  __bf16* aout   = x_bf;  // alias: safe, stages are stream-ordered

  // stage 0: precision conversion (one-time, bandwidth-trivial)
  k_cast_bf16<<<4096, 256, 0, stream>>>(x, x_bf, T * 1024);
  k_transpose_bf16<<<4096, 256, 0, stream>>>(Wqkv, wqkv_t, 3072, 1024);
  k_transpose_bf16<<<2048, 256, 0, stream>>>(Wout, wout_t, 1024, 1024);

  // stage 1: qkv[T,3072] = x[T,1024] @ Wqkv
  k_gemm_bf16out<<<dim3(3072 / 128, T / 256), 256, 0, stream>>>(x_bf, wqkv_t, qkv,
                                                                1024, 3072);
  // stage 2: per-token attention -> aout[T,1024]
  k_attention<<<T / 8, 256, 0, stream>>>(qkv, aout);

  // stage 3: out[T,1024] = aout @ Wout + b
  k_gemm_f32out_bias<<<dim3(1024 / 128, T / 256), 256, 0, stream>>>(aout, wout_t, bout,
                                                                    out, 1024, 1024);
}
